// GCNLayer_30116310679886
// MI455X (gfx1250) — compile-verified, matrix operand order checked by reading, not run on hardware
//
#include <hip/hip_runtime.h>
#include <hip/hip_bf16.h>

typedef __attribute__((ext_vector_type(16))) _Float16 v16h;
typedef __attribute__((ext_vector_type(8)))  float    v8f;

#define N_NODES 100000
#define D_IN    128
#define D_OUT   256

// ---------------------------------------------------------------------------
// Phase 0: A = H   (A lives in workspace; fully initialized every call)
// ---------------------------------------------------------------------------
__global__ void gcn_init_A(const float* __restrict__ H, float* __restrict__ A, int n4) {
    int t = blockIdx.x * blockDim.x + threadIdx.x;
    if (t < n4) {
        reinterpret_cast<float4*>(A)[t] = reinterpret_cast<const float4*>(H)[t];
    }
}

// ---------------------------------------------------------------------------
// Phase 0b: pack W (f32 row-major [128][256]) into per-lane WMMA B fragments.
// Wp layout: [ktile(4)][ntile(16)][lane(32)][half(16)]  -> contiguous 32B/lane
// half i<8  : K = kt*32 + koff + i
// half i>=8 : K = kt*32 + koff + 16 + (i-8)        (koff = (lane>>4)*8)
// col       : ntile*16 + (lane&15)
// ---------------------------------------------------------------------------
__global__ void gcn_pack_W(const float* __restrict__ W, _Float16* __restrict__ Wp) {
    int t = blockIdx.x * blockDim.x + threadIdx.x;    // 0 .. 32767
    if (t >= 4 * 16 * 32 * 16) return;
    int i     = t & 15;
    int lane  = (t >> 4) & 31;
    int ntile = (t >> 9) & 15;
    int ktile = (t >> 13) & 3;
    int koff  = (lane >> 4) * 8;
    int k     = ktile * 32 + koff + ((i < 8) ? i : (8 + i));
    int col   = ntile * 16 + (lane & 15);
    Wp[t] = (_Float16)W[k * D_OUT + col];
}

// ---------------------------------------------------------------------------
// Phase 1: scatter-add messages.  thread -> (edge, 4-feature chunk)
// L2-resident (H + A = 102 MB < 192 MB L2): global_atomic_add_f32 hits L2.
// ---------------------------------------------------------------------------
__global__ void gcn_scatter(const float* __restrict__ H, const int* __restrict__ ei,
                            float* __restrict__ A, int E) {
    unsigned t = blockIdx.x * blockDim.x + threadIdx.x;
    int e = (int)(t >> 5);
    if (e >= E) return;
    int c = (int)(t & 31u) * 4;
    int s = ei[e];
    int d = ei[E + e];
    float4 v = *reinterpret_cast<const float4*>(H + (size_t)s * D_IN + c);
    float* dst = A + (size_t)d * D_IN + c;
    atomicAdd(dst + 0, v.x);
    atomicAdd(dst + 1, v.y);
    atomicAdd(dst + 2, v.z);
    atomicAdd(dst + 3, v.w);
}

// ---------------------------------------------------------------------------
// Phase 2: out = relu(A @ W) via v_wmma_f32_16x16x32_f16.
// Block = 256 threads = 8 waves.  Block covers 2 M-tiles x full 256 cols.
// wave>>2 : which M-tile (0/1); (wave&3)*4 : first of 4 owned N-tiles.
// A fragment reused across 4 WMMAs per k-step.
// ---------------------------------------------------------------------------
__global__ __launch_bounds__(256) void gcn_gemm_relu(const float* __restrict__ A,
                                                     const _Float16* __restrict__ Wp,
                                                     float* __restrict__ out) {
    const int lane  = threadIdx.x & 31;
    const int wave  = threadIdx.x >> 5;
    const int mtile = blockIdx.x * 2 + (wave >> 2);
    const int nt0   = (wave & 3) * 4;          // first N-tile of 4
    const int m     = lane & 15;
    const int koff  = (lane >> 4) * 8;         // 0 or 8

    const float* arow = A + (size_t)(mtile * 16 + m) * D_IN;
    const v16h*  bfr  = reinterpret_cast<const v16h*>(Wp);

    v8f acc0 = {}, acc1 = {}, acc2 = {}, acc3 = {};

#pragma unroll
    for (int kt = 0; kt < 4; ++kt) {
        // ---- A fragment: two contiguous 8-float runs, converted f32->f16 ----
        const float* ap = arow + kt * 32 + koff;
        float4 a0 = *reinterpret_cast<const float4*>(ap);
        float4 a1 = *reinterpret_cast<const float4*>(ap + 4);
        float4 a2 = *reinterpret_cast<const float4*>(ap + 16);
        float4 a3 = *reinterpret_cast<const float4*>(ap + 20);
        v16h a;
        a[0]  = (_Float16)a0.x; a[1]  = (_Float16)a0.y;
        a[2]  = (_Float16)a0.z; a[3]  = (_Float16)a0.w;
        a[4]  = (_Float16)a1.x; a[5]  = (_Float16)a1.y;
        a[6]  = (_Float16)a1.z; a[7]  = (_Float16)a1.w;
        a[8]  = (_Float16)a2.x; a[9]  = (_Float16)a2.y;
        a[10] = (_Float16)a2.z; a[11] = (_Float16)a2.w;
        a[12] = (_Float16)a3.x; a[13] = (_Float16)a3.y;
        a[14] = (_Float16)a3.z; a[15] = (_Float16)a3.w;

        // ---- B fragments: pre-packed, one 32B vector load each ----
        v16h b0 = bfr[((kt * 16 + nt0 + 0) * 32) + lane];
        v16h b1 = bfr[((kt * 16 + nt0 + 1) * 32) + lane];
        v16h b2 = bfr[((kt * 16 + nt0 + 2) * 32) + lane];
        v16h b3 = bfr[((kt * 16 + nt0 + 3) * 32) + lane];

        acc0 = __builtin_amdgcn_wmma_f32_16x16x32_f16(false, a, false, b0,
                                                      (short)0, acc0, false, false);
        acc1 = __builtin_amdgcn_wmma_f32_16x16x32_f16(false, a, false, b1,
                                                      (short)0, acc1, false, false);
        acc2 = __builtin_amdgcn_wmma_f32_16x16x32_f16(false, a, false, b2,
                                                      (short)0, acc2, false, false);
        acc3 = __builtin_amdgcn_wmma_f32_16x16x32_f16(false, a, false, b3,
                                                      (short)0, acc3, false, false);
    }

    // ---- Epilogue: ReLU + store (D layout: VGPR r -> row r + 8*(lane>>4)) ----
    const int n   = lane & 15;
    const int mhi = (lane >> 4) * 8;
#pragma unroll
    for (int r = 0; r < 8; ++r) {
        float* o = out + (size_t)(mtile * 16 + mhi + r) * D_OUT + nt0 * 16 + n;
        o[0]  = fmaxf(acc0[r], 0.0f);
        o[16] = fmaxf(acc1[r], 0.0f);
        o[32] = fmaxf(acc2[r], 0.0f);
        o[48] = fmaxf(acc3[r], 0.0f);
    }
}

// ---------------------------------------------------------------------------
extern "C" void kernel_launch(void* const* d_in, const int* in_sizes, int n_in,
                              void* d_out, int out_size, void* d_ws, size_t ws_size,
                              hipStream_t stream) {
    const float* H  = (const float*)d_in[0];
    const int*   ei = (const int*)d_in[1];       // [2, E] flattened
    const float* W  = (const float*)d_in[2];     // [128, 256]
    float*       out = (float*)d_out;

    const int E = in_sizes[1] / 2;

    // workspace: A (f32, 51.2 MB) | Wp (f16 packed fragments, 64 KB)
    float*    A  = (float*)d_ws;
    _Float16* Wp = (_Float16*)((char*)d_ws + (size_t)N_NODES * D_IN * sizeof(float));

    // Phase 0: A = H
    {
        int n4 = N_NODES * D_IN / 4;                 // 3.2M float4
        gcn_init_A<<<(n4 + 255) / 256, 256, 0, stream>>>(H, A, n4);
    }
    // Phase 0b: pack W into WMMA B-fragment layout
    gcn_pack_W<<<(4 * 16 * 32 * 16 + 255) / 256, 256, 0, stream>>>(W, Wp);

    // Phase 1: scatter-add (32 lanes of float4 per edge)
    {
        long long threads = (long long)E * 32;
        int blocks = (int)((threads + 255) / 256);
        gcn_scatter<<<blocks, 256, 0, stream>>>(H, ei, A, E);
    }

    // Phase 2: WMMA GEMM + ReLU.  100000 rows = 6250 M-tiles = 3125 blocks.
    gcn_gemm_relu<<<(N_NODES / 16) / 2, 256, 0, stream>>>(A, Wp, out);
}
